// AttentionPropagation_32212254720476
// MI455X (gfx1250) — matrix operand order repaired and checked by reading, not compiled, not run on hardware
//
#include <hip/hip_runtime.h>

// ---------------------------------------------------------------------------
// AttentionPropagation for MI455X (gfx1250, wave32, WMMA bf16 16x16x32)
// B=4, C=128, H=4, D=32, N=2048
// Pipeline (all on `stream`, all scratch in d_ws):
//   1) convert weights f32->bf16; transpose-pack x1 -> catT[:, :,0:128] and
//      x2 -> x2T as [B,N,C] bf16 (K-contiguous for WMMA B fragments)
//   2) QKV projection GEMMs (WMMA), producing qT,kT [B,N,C] bf16 and
//      v [B,C,N] bf16
//   3) flash attention, 64-key blocks (online softmax amortized over 64 keys,
//      WMMA QK^T and P*V) -> addvT [B,N,C]
//   4) wm GEMM -> catT[:, :,128:256]
//   5) wc1 GEMM + BatchNorm(inference) + ReLU -> h1T [B,N,2C] bf16
//   6) wc2 GEMM + bias + residual(x1) -> d_out f32 [B,C,N]
// ---------------------------------------------------------------------------

typedef __attribute__((ext_vector_type(16))) __bf16 v16bf;
typedef __attribute__((ext_vector_type(8)))  float  v8f;
typedef __attribute__((ext_vector_type(4)))  unsigned int v4u;

static constexpr int Bb = 4;
static constexpr int Cc = 128;
static constexpr int Hh = 4;
static constexpr int Dd = 32;
static constexpr int Nn = 2048;
static constexpr float BN_EPS = 1e-5f;
static constexpr float SCALE  = 0.17677669529663687f; // 1/sqrt(32)

// ------------------------- WMMA fragment loaders ---------------------------
// A-tile 16(M) x 32(K) bf16, row-major source W[row][k], ld = row stride.
// Contiguous k order used here: lane = (m = lane&15, khalf = lane>>4); VGPR i holds K pair
//   k = khalf*16 + 2*i
__device__ inline v16bf load_a16x32(const __bf16* __restrict__ W, int row0,
                                    int ld, int k0, int lane) {
  int r = lane & 15, kh = lane >> 4;
  union { unsigned int u[8]; v16bf v; } a;
  const __bf16* base = W + (size_t)(row0 + r) * (size_t)ld + k0;
#pragma unroll
  for (int i = 0; i < 8; ++i) {
    int k = kh * 16 + 2 * i;
    a.u[i] = *(const unsigned int*)(base + k);
  }
  return a.v;
}

// B-tile 32(K) x 16(Ncol) bf16 from K-contiguous storage XT[col][k], ld = K stride.
// Contiguous k order used here: lane = (col = lane&15, khalf = lane>>4); VGPR i holds pair
//   k = khalf*16 + 2*i
__device__ inline v16bf load_b32x16(const __bf16* __restrict__ XT, int col0,
                                    size_t ld, int k0, int lane) {
  int c = lane & 15, kh = lane >> 4;
  union { unsigned int u[8]; v16bf v; } b;
  const __bf16* base = XT + (size_t)(col0 + c) * ld + (size_t)(k0 + kh * 16);
#pragma unroll
  for (int i = 0; i < 8; ++i) b.u[i] = *(const unsigned int*)(base + 2 * i);
  return b.v;
}

// One 16x16 f32 tile of  Y = W[row0:+16, :K] * XT[col0:+16, :K]^T
__device__ inline v8f gemm_tile(const __bf16* __restrict__ W,
                                const __bf16* __restrict__ XT,
                                int row0, int col0, int K, size_t ldx, int lane) {
  v8f acc = {};
  for (int k0 = 0; k0 < K; k0 += 32) {
    v16bf a = load_a16x32(W, row0, K, k0, lane);
    v16bf b = load_b32x16(XT, col0, ldx, k0, lane);
    acc = __builtin_amdgcn_wmma_f32_16x16x32_bf16(false, a, false, b,
                                                  (short)0, acc, false, false);
  }
  return acc;
}

// pack 8 f32 -> 8 bf16 -> one 16B store (dst 16B aligned)
__device__ inline void store8_bf16(__bf16* dst, const float* y) {
  union { __bf16 h[8]; v4u u; } t;
#pragma unroll
  for (int r = 0; r < 8; ++r) t.h[r] = (__bf16)y[r];
  *(v4u*)dst = t.u;
}

// ------------------------------ pack kernels -------------------------------
__global__ void convert_w_kernel(const float* __restrict__ s,
                                 __bf16* __restrict__ d, int count) {
  int i = blockIdx.x * blockDim.x + threadIdx.x;
  if (i < count) d[i] = (__bf16)s[i];
}

// x [B,C,N] f32 -> dst[(b*N+n)*ldo + c] bf16   (K-contiguous transpose)
__global__ void pack_xT_kernel(const float* __restrict__ x,
                               __bf16* __restrict__ dst, int ldo) {
  size_t i = (size_t)blockIdx.x * blockDim.x + threadIdx.x; // over B*C*N
  if (i >= (size_t)Bb * Cc * Nn) return;
  int n = (int)(i % Nn);
  int c = (int)((i / Nn) % Cc);
  int b = (int)(i / ((size_t)Nn * Cc));
  dst[((size_t)b * Nn + n) * (size_t)ldo + c] = (__bf16)x[i];
}

// ------------------------------ QKV GEMM -----------------------------------
// grid (N/16, B, 3), block 256 (8 waves cover o=0..127)
__global__ void gemm_qkv_kernel(const __bf16* __restrict__ catT, // x1T half, ld 256
                                const __bf16* __restrict__ x2T,  // ld 128
                                const __bf16* __restrict__ wq,
                                const __bf16* __restrict__ wk,
                                const __bf16* __restrict__ wv,
                                const float* __restrict__ bq,
                                const float* __restrict__ bk,
                                const float* __restrict__ bv,
                                __bf16* __restrict__ qT,  // [B,N,C]
                                __bf16* __restrict__ kT,  // [B,N,C]
                                __bf16* __restrict__ vB)  // [B,C,N]
{
  int lane = threadIdx.x & 31, w = threadIdx.x >> 5;
  int b = blockIdx.y, which = blockIdx.z;
  int col0 = blockIdx.x * 16;  // n within batch
  int row0 = w * 16;           // output channel
  const __bf16* W   = which == 0 ? wq : (which == 1 ? wk : wv);
  const float* bias = which == 0 ? bq : (which == 1 ? bk : bv);
  const __bf16* XT  = which == 0 ? catT : x2T;
  size_t ldx        = which == 0 ? 256 : 128;
  const __bf16* XTb = XT + (size_t)b * Nn * ldx;

  v8f acc = gemm_tile(W, XTb, row0, col0, Cc, ldx, lane);

  int g = lane >> 4, lc = lane & 15;
  float y[8];
#pragma unroll
  for (int r = 0; r < 8; ++r) y[r] = acc[r] + bias[row0 + r + 8 * g];

  int n = col0 + lc;
  if (which == 2) { // v -> [B,C,N]
#pragma unroll
    for (int r = 0; r < 8; ++r) {
      int o = row0 + r + 8 * g;
      vB[((size_t)b * Cc + o) * Nn + n] = (__bf16)y[r];
    }
  } else {          // qT / kT -> [B,N,C], 8 contiguous channels per lane
    __bf16* OT = (which == 0 ? qT : kT);
    store8_bf16(OT + ((size_t)b * Nn + n) * Cc + row0 + 8 * g, y);
  }
}

// --------------------------- flash attention -------------------------------
// grid (N/128, B*H), block 256. Wave w handles 16 query rows; keys processed
// in blocks of 64 so the online-softmax VALU/shfl bookkeeping is amortized
// over twice as many WMMA MACs (the matrix pipe is the resource to keep fed).
__global__ void attn_kernel(const __bf16* __restrict__ qT,
                            const __bf16* __restrict__ kT,
                            const __bf16* __restrict__ vB,
                            const unsigned char* __restrict__ kvmask,
                            __bf16* __restrict__ addvT) // [B,N,C]
{
  int lane = threadIdx.x & 31, w = threadIdx.x >> 5;
  int bh = blockIdx.y, b = bh >> 2, h = bh & 3;
  int n0 = blockIdx.x * 128 + w * 16;
  int g = lane >> 4, lc = lane & 15;

  __shared__ __align__(16) __bf16 Plds[8][16][64];

  // A fragment of Q: 16 rows (n) x 32 cols (d) — constant for whole loop
  union { unsigned int u[8]; v16bf v; } aq;
  {
    const __bf16* base = qT + ((size_t)b * Nn + n0 + lc) * Cc + h * Dd;
#pragma unroll
    for (int i = 0; i < 8; ++i) {
      int k = g * 16 + 2 * i;
      aq.u[i] = *(const unsigned int*)(base + k);
    }
  }

  float mrow[8], lrow[8];
#pragma unroll
  for (int r = 0; r < 8; ++r) { mrow[r] = -1e30f; lrow[r] = 0.0f; }
  v8f acc0 = {}, acc1 = {};

  const __bf16* kbase = kT + ((size_t)b * Nn + lc) * Cc + h * Dd + g * 16;
  const __bf16* vbase = vB + ((size_t)b * Cc + h * Dd + lc) * Nn + g * 16;

  for (int m0 = 0; m0 < Nn; m0 += 64) {
    // prefetch next K/V block into the cache hierarchy (global_prefetch_b8)
    __builtin_prefetch(kbase + (size_t)(m0 + 64) * Cc, 0, 1);
    __builtin_prefetch(vbase + m0 + 64, 0, 1);

    // K fragments: four B-tiles 32(d) x 16(m) covering keys m0..m0+63
    union { unsigned int u[8]; v16bf v; } bk[4];
#pragma unroll
    for (int t = 0; t < 4; ++t) {
      const __bf16* base = kbase + (size_t)(m0 + 16 * t) * Cc;
#pragma unroll
      for (int i = 0; i < 8; ++i)
        bk[t].u[i] = *(const unsigned int*)(base + 2 * i);
    }
    v8f z = {};
    v8f s0 = __builtin_amdgcn_wmma_f32_16x16x32_bf16(false, aq.v, false, bk[0].v,
                                                     (short)0, z, false, false);
    v8f s1 = __builtin_amdgcn_wmma_f32_16x16x32_bf16(false, aq.v, false, bk[1].v,
                                                     (short)0, z, false, false);
    v8f s2 = __builtin_amdgcn_wmma_f32_16x16x32_bf16(false, aq.v, false, bk[2].v,
                                                     (short)0, z, false, false);
    v8f s3 = __builtin_amdgcn_wmma_f32_16x16x32_bf16(false, aq.v, false, bk[3].v,
                                                     (short)0, z, false, false);

    bool ok[4];
#pragma unroll
    for (int t = 0; t < 4; ++t)
      ok[t] = kvmask[(size_t)b * Nn + m0 + 16 * t + lc] != 0;

#pragma unroll
    for (int r = 0; r < 8; ++r) {
      // masked_fill(-1e6) before scaling, matching reference
      float a0 = (ok[0] ? s0[r] : -1e6f) * SCALE;
      float a1 = (ok[1] ? s1[r] : -1e6f) * SCALE;
      float a2 = (ok[2] ? s2[r] : -1e6f) * SCALE;
      float a3 = (ok[3] ? s3[r] : -1e6f) * SCALE;
      // row max across the 16 lanes of this half-wave (rows r / r+8)
      float t = fmaxf(fmaxf(a0, a1), fmaxf(a2, a3));
      for (int mm = 1; mm < 16; mm <<= 1) t = fmaxf(t, __shfl_xor(t, mm, 32));
      float mnew  = fmaxf(mrow[r], t);
      float alpha = __expf(mrow[r] - mnew);
      float e0 = __expf(a0 - mnew);
      float e1 = __expf(a1 - mnew);
      float e2 = __expf(a2 - mnew);
      float e3 = __expf(a3 - mnew);
      float rs = (e0 + e1) + (e2 + e3);
      for (int mm = 1; mm < 16; mm <<= 1) rs += __shfl_xor(rs, mm, 32);
      lrow[r] = lrow[r] * alpha + rs;
      mrow[r] = mnew;
      acc0[r] *= alpha;
      acc1[r] *= alpha;
      // D-layout -> LDS row-major P tile (per-wave private slab)
      Plds[w][r + 8 * g][lc]      = (__bf16)e0;
      Plds[w][r + 8 * g][lc + 16] = (__bf16)e1;
      Plds[w][r + 8 * g][lc + 32] = (__bf16)e2;
      Plds[w][r + 8 * g][lc + 48] = (__bf16)e3;
    }
    asm volatile("s_wait_dscnt 0" ::: "memory"); // LDS store->load, same wave

#pragma unroll
    for (int half = 0; half < 2; ++half) {
      // A fragment of P: 16(n) x 32(m) for keys m0+32*half ..
      union { unsigned int u[8]; v16bf v; } ap;
#pragma unroll
      for (int i = 0; i < 8; ++i) {
        int k = g * 16 + 2 * i;
        ap.u[i] = *(const unsigned int*)&Plds[w][lc][32 * half + k];
      }
      // V fragments: B-tiles 32(m) x 16(d), from v [B,C,N] (m contiguous)
      union { unsigned int u[8]; v16bf v; } bv0, bv1;
      {
        const __bf16* base0 = vbase + m0 + 32 * half;
        const __bf16* base1 = base0 + (size_t)16 * Nn;
#pragma unroll
        for (int i = 0; i < 8; ++i) {
          bv0.u[i] = *(const unsigned int*)(base0 + 2 * i);
          bv1.u[i] = *(const unsigned int*)(base1 + 2 * i);
        }
      }
      acc0 = __builtin_amdgcn_wmma_f32_16x16x32_bf16(false, ap.v, false, bv0.v,
                                                     (short)0, acc0, false, false);
      acc1 = __builtin_amdgcn_wmma_f32_16x16x32_bf16(false, ap.v, false, bv1.v,
                                                     (short)0, acc1, false, false);
    }
  }

  // out[n, d] = acc / l   ->  addvT[(b*N+n)*C + h*32 + d]
#pragma unroll
  for (int r = 0; r < 8; ++r) {
    float inv = 1.0f / lrow[r];
    int n = n0 + r + 8 * g;
    __bf16* dst = addvT + ((size_t)b * Nn + n) * Cc + h * Dd;
    dst[lc]      = (__bf16)(acc0[r] * inv);
    dst[lc + 16] = (__bf16)(acc1[r] * inv);
  }
}

// ------------------------------ wm GEMM ------------------------------------
// grid (N/16, B), writes second half of catT
__global__ void gemm_m_kernel(const __bf16* __restrict__ addvT,
                              const __bf16* __restrict__ wm,
                              const float* __restrict__ bm,
                              __bf16* __restrict__ catT) {
  int lane = threadIdx.x & 31, w = threadIdx.x >> 5;
  int b = blockIdx.y, col0 = blockIdx.x * 16, row0 = w * 16;
  const __bf16* XTb = addvT + (size_t)b * Nn * Cc;
  v8f acc = gemm_tile(wm, XTb, row0, col0, Cc, Cc, lane);
  int g = lane >> 4, lc = lane & 15;
  float y[8];
#pragma unroll
  for (int r = 0; r < 8; ++r) y[r] = acc[r] + bm[row0 + r + 8 * g];
  int n = col0 + lc;
  store8_bf16(catT + ((size_t)b * Nn + n) * 256 + 128 + row0 + 8 * g, y);
}

// ---------------------- wc1 GEMM + BN(inference) + ReLU --------------------
// grid (N/16, B, 2): z selects 128-row half of M=256
__global__ void gemm_c1_kernel(const __bf16* __restrict__ catT,
                               const __bf16* __restrict__ wc1,
                               const float* __restrict__ bc1,
                               const float* __restrict__ gamma,
                               const float* __restrict__ beta,
                               const float* __restrict__ mean,
                               const float* __restrict__ var,
                               __bf16* __restrict__ h1T) {
  int lane = threadIdx.x & 31, w = threadIdx.x >> 5;
  int b = blockIdx.y, col0 = blockIdx.x * 16;
  int row0 = blockIdx.z * 128 + w * 16;
  const __bf16* XTb = catT + (size_t)b * Nn * 256;
  v8f acc = gemm_tile(wc1, XTb, row0, col0, 256, 256, lane);
  int g = lane >> 4, lc = lane & 15;
  float y[8];
#pragma unroll
  for (int r = 0; r < 8; ++r) {
    int o = row0 + r + 8 * g;
    float v0 = acc[r] + bc1[o];
    v0 = (v0 - mean[o]) * __frsqrt_rn(var[o] + BN_EPS) * gamma[o] + beta[o];
    y[r] = fmaxf(v0, 0.0f);
  }
  int n = col0 + lc;
  store8_bf16(h1T + ((size_t)b * Nn + n) * 256 + row0 + 8 * g, y);
}

// ----------------------- wc2 GEMM + bias + residual ------------------------
// grid (N/16, B): out f32 [B,C,N] = x1 + wc2*h1 + bc2
__global__ void gemm_c2_kernel(const __bf16* __restrict__ h1T,
                               const __bf16* __restrict__ wc2,
                               const float* __restrict__ bc2,
                               const float* __restrict__ x1,
                               float* __restrict__ out) {
  int lane = threadIdx.x & 31, w = threadIdx.x >> 5;
  int b = blockIdx.y, col0 = blockIdx.x * 16, row0 = w * 16;
  const __bf16* XTb = h1T + (size_t)b * Nn * 256;
  v8f acc = gemm_tile(wc2, XTb, row0, col0, 256, 256, lane);
  int g = lane >> 4, lc = lane & 15;
  int n = col0 + lc;
#pragma unroll
  for (int r = 0; r < 8; ++r) {
    int o = row0 + r + 8 * g;
    size_t idx = ((size_t)b * Cc + o) * Nn + n;
    out[idx] = x1[idx] + acc[r] + bc2[o];
  }
}

// ------------------------------- launcher ----------------------------------
extern "C" void kernel_launch(void* const* d_in, const int* in_sizes, int n_in,
                              void* d_out, int out_size, void* d_ws, size_t ws_size,
                              hipStream_t stream) {
  (void)in_sizes; (void)n_in; (void)out_size; (void)ws_size;
  const float* x1 = (const float*)d_in[0];
  const float* x2 = (const float*)d_in[1];
  const unsigned char* kvmask = (const unsigned char*)d_in[2];
  const float* wq  = (const float*)d_in[3];
  const float* bq  = (const float*)d_in[4];
  const float* wk  = (const float*)d_in[5];
  const float* bk  = (const float*)d_in[6];
  const float* wv  = (const float*)d_in[7];
  const float* bv  = (const float*)d_in[8];
  const float* wm  = (const float*)d_in[9];
  const float* bm  = (const float*)d_in[10];
  const float* wc1 = (const float*)d_in[11];
  const float* bc1 = (const float*)d_in[12];
  const float* bng = (const float*)d_in[13];
  const float* bnb = (const float*)d_in[14];
  const float* bnm = (const float*)d_in[15];
  const float* bnv = (const float*)d_in[16];
  const float* wc2 = (const float*)d_in[17];
  const float* bc2 = (const float*)d_in[18];

  char* ws = (char*)d_ws;
  size_t off = 0;
  auto take = [&](size_t bytes) { size_t o = off; off += (bytes + 255) & ~size_t(255); return o; };
  __bf16* catT  = (__bf16*)(ws + take((size_t)Bb * Nn * 256 * 2)); // [B,N,2C]
  __bf16* x2T   = (__bf16*)(ws + take((size_t)Bb * Nn * Cc * 2));
  __bf16* qT    = (__bf16*)(ws + take((size_t)Bb * Nn * Cc * 2));
  __bf16* kT    = (__bf16*)(ws + take((size_t)Bb * Nn * Cc * 2));
  __bf16* vB    = (__bf16*)(ws + take((size_t)Bb * Cc * Nn * 2));
  __bf16* addvT = (__bf16*)(ws + take((size_t)Bb * Nn * Cc * 2));
  __bf16* h1T   = (__bf16*)(ws + take((size_t)Bb * Nn * 256 * 2));
  __bf16* wq_b  = (__bf16*)(ws + take((size_t)Cc * Cc * 2));
  __bf16* wk_b  = (__bf16*)(ws + take((size_t)Cc * Cc * 2));
  __bf16* wv_b  = (__bf16*)(ws + take((size_t)Cc * Cc * 2));
  __bf16* wm_b  = (__bf16*)(ws + take((size_t)Cc * Cc * 2));
  __bf16* wc1_b = (__bf16*)(ws + take((size_t)256 * 256 * 2));
  __bf16* wc2_b = (__bf16*)(ws + take((size_t)Cc * 256 * 2));

  // 1) weight conversion + input transpose/pack
  convert_w_kernel<<<(Cc * Cc + 255) / 256, 256, 0, stream>>>(wq, wq_b, Cc * Cc);
  convert_w_kernel<<<(Cc * Cc + 255) / 256, 256, 0, stream>>>(wk, wk_b, Cc * Cc);
  convert_w_kernel<<<(Cc * Cc + 255) / 256, 256, 0, stream>>>(wv, wv_b, Cc * Cc);
  convert_w_kernel<<<(Cc * Cc + 255) / 256, 256, 0, stream>>>(wm, wm_b, Cc * Cc);
  convert_w_kernel<<<(256 * 256 + 255) / 256, 256, 0, stream>>>(wc1, wc1_b, 256 * 256);
  convert_w_kernel<<<(Cc * 256 + 255) / 256, 256, 0, stream>>>(wc2, wc2_b, Cc * 256);
  int ntr = (Bb * Cc * Nn + 255) / 256;
  pack_xT_kernel<<<ntr, 256, 0, stream>>>(x1, catT, 256); // first half of catT
  pack_xT_kernel<<<ntr, 256, 0, stream>>>(x2, x2T, 128);

  // 2) QKV projections
  gemm_qkv_kernel<<<dim3(Nn / 16, Bb, 3), 256, 0, stream>>>(
      catT, x2T, wq_b, wk_b, wv_b, bq, bk, bv, qT, kT, vB);

  // 3) flash attention (64-key blocks)
  attn_kernel<<<dim3(Nn / 128, Bb * Hh), 256, 0, stream>>>(qT, kT, vB, kvmask, addvT);

  // 4) wm projection into catT second half
  gemm_m_kernel<<<dim3(Nn / 16, Bb), 256, 0, stream>>>(addvT, wm_b, bm, catT);

  // 5) wc1 + BN + ReLU
  gemm_c1_kernel<<<dim3(Nn / 16, Bb, 2), 256, 0, stream>>>(
      catT, wc1_b, bc1, bng, bnb, bnm, bnv, h1T);

  // 6) wc2 + residual
  gemm_c2_kernel<<<dim3(Nn / 16, Bb), 256, 0, stream>>>(h1T, wc2_b, bc2, x1, (float*)d_out);
}
